// Coloring_Transformer_17016660427420
// MI455X (gfx1250) — matrix operand order repaired and checked by reading, not compiled, not run on hardware
//
#include <hip/hip_runtime.h>
#include <hip/hip_bf16.h>

typedef __attribute__((ext_vector_type(16))) __bf16 v16bf;
typedef __attribute__((ext_vector_type(8)))  float  v8f;

#define NN 20000
#define NE 320000
#define NH 4

// ---------------- conversion / fill ----------------
__global__ void k_f32_to_bf16(const float* __restrict__ s, __bf16* __restrict__ d, int n){
  int i = blockIdx.x*blockDim.x + threadIdx.x;
  if (i < n) d[i] = (__bf16)s[i];
}
// transpose weights to [N,K] so B fragments are k-contiguous per lane
__global__ void k_w_to_bf16_t(const float* __restrict__ W, __bf16* __restrict__ Wt,
                              int Kd, int Ncol){
  int i = blockIdx.x*blockDim.x + threadIdx.x;
  if (i < Kd*Ncol){
    int k = i / Ncol, n = i % Ncol;
    Wt[(size_t)n*Kd + k] = (__bf16)W[i];
  }
}
__global__ void k_fill(float* __restrict__ p, float v, int n){
  int i = blockIdx.x*blockDim.x + threadIdx.x;
  if (i < n) p[i] = v;
}

// ---------------- WMMA GEMM: C[M,Ncol] = A[M,KD]@B[KD,Ncol] + bias ----------------
// A row-major [M,KD] bf16; Bt transposed [Ncol,KD] bf16. One wave per 16x16 tile.
// Fragment layouts (ISA 7.12.2, wave32):
//   A: lane m=lane%16, half=lane/16; frag elems 0..7  -> k = kstep+8*half+i (contig)
//                                     frag elems 8..15 -> k = kstep+16+8*half+(i-8) (contig)
//   B: lane n=lane%16; frag elems in the same k order as A (two 8-bf16 runs in Bt row n)
//   C: VGPR r -> row r+8*half, col n.
template<int KD>
__global__ void k_gemm_wmma(const __bf16* __restrict__ A, const __bf16* __restrict__ Bt,
                            const float* __restrict__ bias, float* __restrict__ C,
                            int M, int Ncol){
  int wave = (int)((blockIdx.x*blockDim.x + threadIdx.x) >> 5);
  int lane = threadIdx.x & 31;
  int tilesN = Ncol >> 4;
  int tiles  = (M >> 4) * tilesN;
  if (wave >= tiles) return;               // uniform per wave: EXEC stays all-ones
  int tm = wave / tilesN, tn = wave % tilesN;
  int row0 = tm << 4, col0 = tn << 4;
  int half = lane >> 4;
  int l16  = lane & 15;
  int m = row0 + l16;
  int n = col0 + l16;
  const __bf16* arow = A  + (size_t)m*KD;
  const __bf16* brow = Bt + (size_t)n*KD;
  unsigned zm = half ? 0u : ~0u;           // branchless zero-pad mask (KD==16 only)

  v8f acc = {};
  #pragma unroll
  for (int kstep = 0; kstep < ((KD+31)/32)*32; kstep += 32){
    union { v16bf v; uint4 q[2]; } af;
    union { v16bf v; uint4 q[2]; } bfr;
    if (KD == 64){
      af.q[0] = *(const uint4*)(arow + kstep + 8*half);
      af.q[1] = *(const uint4*)(arow + kstep + 16 + 8*half);
      bfr.q[0] = *(const uint4*)(brow + kstep + 8*half);
      bfr.q[1] = *(const uint4*)(brow + kstep + 16 + 8*half);
    } else {                               // KD == 16, single k-step, zero-padded to 32
      af.q[0] = *(const uint4*)(arow + 8*half);          // k = 8*half..+7 (valid)
      af.q[1] = make_uint4(0u,0u,0u,0u);                 // k >= 16 -> 0
      // B matches A: elems 0..7 -> k = 8*half..+7 (valid), elems 8..15 -> k >= 16 -> 0
      uint4 b0 = *(const uint4*)(brow + 8*half);
      uint4 b1 = make_uint4(0u,0u,0u,0u);
      (void)zm;
      bfr.q[0] = b0; bfr.q[1] = b1;
    }
    acc = __builtin_amdgcn_wmma_f32_16x16x32_bf16(false, af.v, false, bfr.v,
                                                  (short)0, acc, false, false);
  }
  float bn = bias[n];
  #pragma unroll
  for (int r = 0; r < 8; ++r){
    int mm = row0 + r + (half ? 8 : 0);
    C[(size_t)mm*Ncol + n] = acc[r] + bn;
  }
}

// ---------------- edge attention logits + segment max ----------------
__global__ void k_edge_logits(const float* __restrict__ Q, const float* __restrict__ K,
                              const long long* __restrict__ src, const long long* __restrict__ dst,
                              float* __restrict__ alpha, float* __restrict__ maxb, int Cc){
  int e = (int)((blockIdx.x*blockDim.x + threadIdx.x) >> 5);
  if (e >= NE) return;
  int lane = threadIdx.x & 31;
  int HC  = NH * Cc;
  int per = HC >> 5;                       // floats per lane (8 for C=64, 2 for C=16)
  long long sN = src[e], dN = dst[e];
  const float* q = Q + (size_t)dN * HC;
  const float* k = K + (size_t)sN * HC;
  float s = 0.f;
  for (int j = 0; j < per; ++j){ int idx = lane*per + j; s += q[idx] * k[idx]; }
  s += __shfl_xor(s, 1, 32);               // reduce within 8-lane head group
  s += __shfl_xor(s, 2, 32);
  s += __shfl_xor(s, 4, 32);
  if ((lane & 7) == 0){
    int h = lane >> 3;
    float val = s * rsqrtf((float)Cc);
    alpha[(size_t)e*NH + h] = val;
    atomicMax(maxb + (size_t)dN*NH + h, val);   // global_atomic_max_num_f32
  }
}

// ---------------- exp(alpha - max) + segment sum ----------------
__global__ void k_edge_exp(float* __restrict__ alpha, const long long* __restrict__ dst,
                           const float* __restrict__ maxb, float* __restrict__ den){
  int i = blockIdx.x*blockDim.x + threadIdx.x;
  if (i >= NE*NH) return;
  int e = i >> 2, h = i & 3;
  long long dN = dst[e];
  float ex = __expf(alpha[i] - maxb[(size_t)dN*NH + h]);
  alpha[i] = ex;
  atomicAdd(den + (size_t)dN*NH + h, ex);
}

// ---------------- weighted messages scatter-add ----------------
__global__ void k_edge_msg(const float* __restrict__ V, const float* __restrict__ alpha,
                           const float* __restrict__ den, const long long* __restrict__ src,
                           const long long* __restrict__ dst, float* __restrict__ agg, int Cc){
  int e = (int)((blockIdx.x*blockDim.x + threadIdx.x) >> 5);
  if (e >= NE) return;
  int lane = threadIdx.x & 31;
  int HC  = NH * Cc;
  int per = HC >> 5;
  long long sN = src[e], dN = dst[e];
  int h = lane >> 3;                       // lane's head (per*8 == Cc)
  float a = alpha[(size_t)e*NH + h] / (den[(size_t)dN*NH + h] + 1e-16f);
  const float* v  = V   + (size_t)sN * HC;
  float*       ag = agg + (size_t)dN * HC;
  for (int j = 0; j < per; ++j){
    int idx = lane*per + j;
    atomicAdd(ag + idx, v[idx] * a);       // global_atomic_add_f32
  }
}

// ---------------- head mean + skip + relu ----------------
__global__ void k_combine(const float* __restrict__ agg, const float* __restrict__ S,
                          float* __restrict__ out, int Cc){
  int i = blockIdx.x*blockDim.x + threadIdx.x;
  if (i >= NN*Cc) return;
  int n = i / Cc, d = i % Cc;
  int HC = NH * Cc;
  float s = 0.f;
  #pragma unroll
  for (int h = 0; h < NH; ++h) s += agg[(size_t)n*HC + h*Cc + d];
  s = s * (1.0f/NH) + S[i];
  out[i] = fmaxf(s, 0.f);
}

// ---------------- softmax over first 5 cols (width 64), writes x5 output ----------------
__global__ void k_mid_softmax(float* __restrict__ h, float* __restrict__ x5){
  int n = blockIdx.x*blockDim.x + threadIdx.x;
  if (n >= NN) return;
  float v[5]; float m = -1e30f;
  #pragma unroll
  for (int j = 0; j < 5; ++j){ v[j] = h[(size_t)n*64 + j]; m = fmaxf(m, v[j]); }
  float sden = 0.f;
  #pragma unroll
  for (int j = 0; j < 5; ++j){ v[j] = __expf(v[j] - m); sden += v[j]; }
  #pragma unroll
  for (int j = 0; j < 5; ++j){
    float o = v[j] / sden;
    x5[(size_t)n*5 + j] = o;
    h[(size_t)n*64 + j] = o;
  }
}

// ---------------- final softmax over 16 cols ----------------
__global__ void k_final_softmax(const float* __restrict__ h, float* __restrict__ out){
  int n = blockIdx.x*blockDim.x + threadIdx.x;
  if (n >= NN) return;
  float v[16]; float m = -1e30f;
  #pragma unroll
  for (int j = 0; j < 16; ++j){ v[j] = h[(size_t)n*16 + j]; m = fmaxf(m, v[j]); }
  float sden = 0.f;
  #pragma unroll
  for (int j = 0; j < 16; ++j){ v[j] = __expf(v[j] - m); sden += v[j]; }
  #pragma unroll
  for (int j = 0; j < 16; ++j) out[(size_t)n*16 + j] = v[j] / sden;
}

// =====================================================================
static inline int cdiv(int a, int b){ return (a + b - 1) / b; }

extern "C" void kernel_launch(void* const* d_in, const int* in_sizes, int n_in,
                              void* d_out, int out_size, void* d_ws, size_t ws_size,
                              hipStream_t stream){
  const float*     x  = (const float*)d_in[0];
  const long long* ei = (const long long*)d_in[1];   // int64 [2, NE]
  const long long* src = ei;
  const long long* dst = ei + NE;

  // workspace carve-out
  size_t off = 0;
  auto alloc = [&](size_t bytes)->void*{
    void* p = (char*)d_ws + off;
    off += (bytes + 255) & ~(size_t)255;
    return p;
  };
  __bf16* bfA  = (__bf16*)alloc((size_t)NN * 64 * 2);   // activations, row-major [M,K]
  __bf16* bfWt = (__bf16*)alloc((size_t)64 * 256 * 2);  // weights transposed [N,K] (+slack)
  float*  Qb   = (float*)alloc((size_t)NN * 256 * 4);
  float*  Kb   = (float*)alloc((size_t)NN * 256 * 4);
  float*  Vb   = (float*)alloc((size_t)NN * 256 * 4);
  float*  Sb   = (float*)alloc((size_t)NN * 64 * 4);
  float*  alp  = (float*)alloc((size_t)NE * NH * 4);
  float*  maxb = (float*)alloc((size_t)NN * NH * 4);
  float*  den  = (float*)alloc((size_t)NN * NH * 4);
  float*  agg  = (float*)alloc((size_t)NN * 256 * 4);
  float*  hb0  = (float*)alloc((size_t)NN * 64 * 4);
  float*  hb1  = (float*)alloc((size_t)NN * 64 * 4);
  float*  hbuf[2] = { hb0, hb1 };
  (void)ws_size; (void)in_sizes; (void)n_in; (void)out_size;

  const int dins[6]  = {16, 64, 64, 64, 64, 64};
  const int couts[6] = {64, 64, 64, 64, 64, 16};

  const float* cur = x;
  float* nxt = nullptr;

  for (int L = 0; L < 6; ++L){
    int din = dins[L], Cc = couts[L], HC = NH * Cc;
    const float* Wq = (const float*)d_in[2 + L*8 + 0];
    const float* bq = (const float*)d_in[2 + L*8 + 1];
    const float* Wk = (const float*)d_in[2 + L*8 + 2];
    const float* bk = (const float*)d_in[2 + L*8 + 3];
    const float* Wv = (const float*)d_in[2 + L*8 + 4];
    const float* bv = (const float*)d_in[2 + L*8 + 5];
    const float* Ws = (const float*)d_in[2 + L*8 + 6];
    const float* bs = (const float*)d_in[2 + L*8 + 7];

    // activations -> bf16 row-major
    k_f32_to_bf16<<<cdiv(NN*din,256),256,0,stream>>>(cur, bfA, NN*din);

    // 4 GEMMs: Q, K, V (Ncol=HC) and skip S (Ncol=Cc)
    const float* Wmats[4]  = { Wq, Wk, Wv, Ws };
    const float* bvecs[4]  = { bq, bk, bv, bs };
    float*       outs[4]   = { Qb, Kb, Vb, Sb };
    int          ncols[4]  = { HC, HC, HC, Cc };
    for (int g = 0; g < 4; ++g){
      int ncol = ncols[g];
      k_w_to_bf16_t<<<cdiv(din*ncol,256),256,0,stream>>>(Wmats[g], bfWt, din, ncol);
      int tiles = (NN/16) * (ncol/16);
      if (din == 64)
        k_gemm_wmma<64><<<cdiv(tiles,8),256,0,stream>>>(bfA, bfWt, bvecs[g], outs[g], NN, ncol);
      else
        k_gemm_wmma<16><<<cdiv(tiles,8),256,0,stream>>>(bfA, bfWt, bvecs[g], outs[g], NN, ncol);
    }

    // reset segment buffers
    k_fill<<<cdiv(NN*NH,256),256,0,stream>>>(maxb, -1e30f, NN*NH);
    k_fill<<<cdiv(NN*NH,256),256,0,stream>>>(den, 0.f, NN*NH);
    k_fill<<<cdiv(NN*HC,256),256,0,stream>>>(agg, 0.f, NN*HC);

    // edge phase: logits+max, exp+sum, weighted scatter
    k_edge_logits<<<cdiv(NE*32,256),256,0,stream>>>(Qb, Kb, src, dst, alp, maxb, Cc);
    k_edge_exp   <<<cdiv(NE*NH,256),256,0,stream>>>(alp, dst, maxb, den);
    k_edge_msg   <<<cdiv(NE*32,256),256,0,stream>>>(Vb, alp, den, src, dst, agg, Cc);

    // head mean + skip + relu
    nxt = hbuf[L & 1];
    k_combine<<<cdiv(NN*Cc,256),256,0,stream>>>(agg, Sb, nxt, Cc);

    if (L == 2)
      k_mid_softmax<<<cdiv(NN,256),256,0,stream>>>(nxt, (float*)d_out);

    cur = nxt;
  }

  // final softmax over 16 cols -> second output
  k_final_softmax<<<cdiv(NN,256),256,0,stream>>>(cur, (float*)d_out + (size_t)NN*5);
}
